// GroupGenerator_23304492548376
// MI455X (gfx1250) — matrix operand order repaired and checked by reading, not compiled, not run on hardware
//
#include <hip/hip_runtime.h>
#include <hip/hip_bf16.h>

// ---- CDNA5 WMMA vector types (gfx1250, wave32) ----
typedef __attribute__((ext_vector_type(16))) __bf16 v16bf;
typedef __attribute__((ext_vector_type(8)))  float  v8f;

#define N_PED 2048
#define SEQ   8
#define TH_   1.0f

// bf16 truncation of a float's top 16 bits (exact for 1.0f and NaN).
static __device__ __forceinline__ __bf16 trunc_bf16(float f) {
    return __builtin_bit_cast(__bf16, (unsigned short)(__float_as_uint(f) >> 16));
}
static __device__ __forceinline__ float trunc_bf16_as_f32(float f) {
    return __uint_as_float(__float_as_uint(f) & 0xFFFF0000u);
}
static __device__ __forceinline__ __bf16 bf16_bits(unsigned short u) {
    return __builtin_bit_cast(__bf16, u);
}

// ---------------------------------------------------------------------------
// Kernel 1: dist[i][j] = mean_t sqrt((x_ti - x_tj)^2 + (y_ti - y_tj)^2)
// v_abs layout (1,2,T,N) flat: x_t,i = vabs[t*N + i], y_t,i = vabs[(T+t)*N + i]
// Source array is 128KB -> fully cache resident; plain loads are fine.
// ---------------------------------------------------------------------------
__global__ __launch_bounds__(256) void dist_kernel(const float* __restrict__ vabs,
                                                   float* __restrict__ dist) {
    const int idx = blockIdx.x * 256 + threadIdx.x;   // 0 .. N*N-1
    const int i = idx >> 11;
    const int j = idx & (N_PED - 1);
    float acc = 0.0f;
#pragma unroll
    for (int t = 0; t < SEQ; ++t) {
        float dx = vabs[t * N_PED + i]         - vabs[t * N_PED + j];
        float dy = vabs[(SEQ + t) * N_PED + i] - vabs[(SEQ + t) * N_PED + j];
        acc += sqrtf(dx * dx + dy * dy);       // s==0 -> sqrtf(0)==0, matches ref
    }
    dist[idx] = acc * (1.0f / SEQ);
}

// ---------------------------------------------------------------------------
// Kernel 2: sequential group relabeling (faithful to the reference scan).
// Single workgroup; labels + row bitmask + rank all in LDS.
// ---------------------------------------------------------------------------
__global__ __launch_bounds__(1024) void label_kernel(const float* __restrict__ dist,
                                                     float* __restrict__ out_idx) {
    __shared__ int          labels[N_PED];
    __shared__ int          rank  [N_PED];
    __shared__ unsigned int rowmask[N_PED / 32];
    __shared__ int          s_ck;

    const int tid = threadIdx.x;
    for (int i = tid; i < N_PED; i += 1024) labels[i] = i;
    __syncthreads();

    for (int r = 0; r < N_PED; ++r) {
        // clear row state
        for (int w = tid; w < N_PED / 32; w += 1024) rowmask[w] = 0u;
        if (tid == 0) s_ck = -1;
        __syncthreads();

        // edges of row r: strict lower triangle, dist <= TH
        for (int c = tid; c < N_PED; c += 1024) {
            if (c < r && dist[r * N_PED + c] <= TH_) {
                atomicOr(&rowmask[c >> 5], 1u << (c & 31));
                atomicMax(&s_ck, c);
            }
        }
        __syncthreads();

        const int ck = s_ck;
        const int lr = labels[r];
        const int l0 = labels[tid];
        const int l1 = labels[tid + 1024];
        __syncthreads();

        if (ck >= 0) {
            bool in0 = (l0 == lr) ||
                       (((rowmask[l0 >> 5] >> (l0 & 31)) & 1u) && (l0 != ck));
            bool in1 = (l1 == lr) ||
                       (((rowmask[l1 >> 5] >> (l1 & 31)) & 1u) && (l1 != ck));
            if (in0) labels[tid]        = ck;
            if (in1) labels[tid + 1024] = ck;
        }
        __syncthreads();
    }

    // rank-remap surviving labels to consecutive ids
    for (int i = tid; i < N_PED; i += 1024) rank[i] = 0;
    __syncthreads();
    for (int i = tid; i < N_PED; i += 1024) rank[labels[i]] = 1;
    __syncthreads();
    if (tid == 0) {
        int s = 0;
        for (int i = 0; i < N_PED; ++i) { s += rank[i]; rank[i] = s - 1; }
    }
    __syncthreads();
    for (int i = tid; i < N_PED; i += 1024)
        out_idx[i] = (float)rank[labels[i]];
}

// ---------------------------------------------------------------------------
// Kernel 3: v_soft = A(16x2048) * B(2048x2048) via v_wmma_f32_16x16x32_bf16.
// B[i][j] = sig/sum(sig,axis=0) over a size-1 axis = sig/sig, which is exactly
// 1.0f when sig > 0 and NaN when sig == 0. sig == 0 iff exp(-z) underflows to
// zero (the /(1+ez) step cannot zero a positive subnormal under RNE), so each
// B element is one v_exp_f32 + compare + select of the bf16 bit pattern:
// 0x3F80 (1.0) or 0x7FC0 (NaN). No divisions in the inner loop.
// A is split into 3 truncated-bf16 terms (hi/mid/lo, residual <= 2^-24|a|)
// -> f32-accurate product from bf16 WMMA. One wave per 16-column tile;
// EXEC all-ones around the WMMAs.
// ---------------------------------------------------------------------------
__global__ __launch_bounds__(32) void softgroup_wmma_kernel(const float* __restrict__ v,
                                                            const float* __restrict__ dist,
                                                            float* __restrict__ out) {
    const int lane  = threadIdx.x;     // 0..31
    const int h     = lane >> 4;       // lane half (ISA 16-bit fragment layout)
    const int mn    = lane & 15;       // M index (A/D) / N index (B/D)
    const int jbase = blockIdx.x * 16; // output column tile

    v8f acc = {};                      // f32 accumulator (C/D)

    for (int kt = 0; kt < N_PED / 32; ++kt) {
        const int K0 = kt * 32;

        // ---- A fragment: rows m = (c*T + t), 16x32 bf16 layout per ISA:
        // element e of lane (h,mn):  K = K0 + 8h + e (+8 if e>=8)
        // => two runs of 8 consecutive floats -> four aligned float4 loads.
        const float* arow = v + mn * N_PED + K0 + 8 * h;
        const float4 a0 = *(const float4*)(arow);
        const float4 a1 = *(const float4*)(arow + 4);
        const float4 a2 = *(const float4*)(arow + 16);
        const float4 a3 = *(const float4*)(arow + 20);
        const float av[16] = {a0.x, a0.y, a0.z, a0.w, a1.x, a1.y, a1.z, a1.w,
                              a2.x, a2.y, a2.z, a2.w, a3.x, a3.y, a3.z, a3.w};

        v16bf ahi, amid, alo;
#pragma unroll
        for (int e = 0; e < 16; ++e) {
            const float a   = av[e];
            const float fhi = trunc_bf16_as_f32(a);
            const float r1  = a - fhi;
            const float fm  = trunc_bf16_as_f32(r1);
            const float r2  = r1 - fm;
            ahi[e]  = trunc_bf16(a);
            amid[e] = trunc_bf16(r1);
            alo[e]  = trunc_bf16(r2);
        }

        // ---- B fragment: 32x16 bf16 layout per ISA: element e -> K = 8h + e (+8 if e>=8)
        v16bf b;
#pragma unroll
        for (int e = 0; e < 16; ++e) {
            const float d  = dist[(K0 + 8 * h + e + (e & 8)) * N_PED + (jbase + mn)];
            const float z  = (d - TH_) * 10.0f;          // (d - TH)/tau
            const float ez = __expf(-z);                 // no overflow: z >= -10
            // sig = ez/(1+ez): zero iff ez == 0 -> sig/sig is 1.0 or NaN
            b[e] = bf16_bits((ez == 0.0f) ? (unsigned short)0x7FC0u
                                          : (unsigned short)0x3F80u);
        }

        // bf16x3 emulated-f32 GEMM: D += (A_hi + A_mid + A_lo) * B
        acc = __builtin_amdgcn_wmma_f32_16x16x32_bf16(false, ahi,  false, b,
                                                      (short)0, acc, false, false);
        acc = __builtin_amdgcn_wmma_f32_16x16x32_bf16(false, amid, false, b,
                                                      (short)0, acc, false, false);
        acc = __builtin_amdgcn_wmma_f32_16x16x32_bf16(false, alo,  false, b,
                                                      (short)0, acc, false, false);
    }

    // D layout: element r of lane (h,mn): M = r + 8h, N = jbase + mn
    // out = stop_gradient(v0 - v_soft) + v_soft  ->  (v - vs) + vs in f32
#pragma unroll
    for (int r = 0; r < 8; ++r) {
        const int m   = r + 8 * h;
        const int col = jbase + mn;
        const float vv = v[m * N_PED + col];
        const float vs = acc[r];
        out[m * N_PED + col] = (vv - vs) + vs;
    }
}

// ---------------------------------------------------------------------------
extern "C" void kernel_launch(void* const* d_in, const int* in_sizes, int n_in,
                              void* d_out, int out_size, void* d_ws, size_t ws_size,
                              hipStream_t stream) {
    (void)in_sizes; (void)n_in; (void)out_size; (void)ws_size;

    const float* v    = (const float*)d_in[0];   // (1,2,8,2048) f32
    const float* vabs = (const float*)d_in[1];   // (1,2,8,2048) f32
    float* out  = (float*)d_out;                 // 2*8*2048 floats, then 2048 indices
    float* dist = (float*)d_ws;                  // 2048*2048 f32 scratch (16 MiB)

    dist_kernel<<<(N_PED * N_PED) / 256, 256, 0, stream>>>(vabs, dist);
    softgroup_wmma_kernel<<<N_PED / 16, 32, 0, stream>>>(v, dist, out);
    label_kernel<<<1, 1024, 0, stream>>>(dist, out + 2 * SEQ * N_PED);
}
